// _GlobalWorkspace_44178033606668
// MI455X (gfx1250) — matrix operand order repaired and checked by reading, not compiled, not run on hardware
//
#include <hip/hip_runtime.h>

// ---------------------------------------------------------------------------
// MI455X (gfx1250): bf16 WMMA GEMM pipeline fed by GLOBAL_LOAD_ASYNC_TO_LDS
// (CDNA5 async copy, ASYNCcnt) with double-buffered LDS slabs, plus small
// fused attention/softmax/LN kernels.  B=8192, K=16, D=512, S=4.
//
// Roofline: ~352 GFLOP total, ~0.5 GB mandatory HBM traffic (22 us floor at
// 23.3 TB/s) -> compute bound; all GEMM FLOPs go through
// v_wmma_f32_16x16x32_bf16 (f32 accumulate).  Weights (17 MB) live in the
// 192 MB L2; batched-GEMM decomposition amortizes them across 128-row tiles.
// ---------------------------------------------------------------------------

typedef __bf16 bf16;
typedef __attribute__((ext_vector_type(16))) __bf16 v16bf;
typedef __attribute__((ext_vector_type(4)))  __bf16 v4bf;
typedef __attribute__((ext_vector_type(8)))  float  v8f;

__device__ __forceinline__ float b2f(bf16 x) { return (float)x; }
__device__ __forceinline__ bf16  f2b(float x) { return (bf16)x; }

// CDNA5 async DMA: copy 16B from global directly into LDS (ASYNCcnt tracked).
// ISA low-32 bits of a flat shared address == LDS offset (aperture rule).
__device__ __forceinline__ void async_b128(unsigned lds_off, const void* gsrc) {
  asm volatile("global_load_async_to_lds_b128 %0, %1, off"
               :: "v"(lds_off), "v"(gsrc) : "memory");
}
__device__ __forceinline__ void wait_async0() {
  asm volatile("s_wait_asynccnt 0" ::: "memory");
}
__device__ __forceinline__ unsigned lds_addr(const void* p) {
  return (unsigned)(unsigned long long)p;
}

// LDS tiles use element stride 40 (80 B rows): 16B-aligned fragment chunks,
// conflict-free stride-20-dword bank pattern.
union FragV { v16bf v; uint4 q[2]; };

// A fragment (16x32 bf16), LDS row-major [row][k], stride 40.
__device__ __forceinline__ v16bf load_frag_a40(const bf16* As, int mrow_base, int lane) {
  int l = lane & 15, half = lane >> 4;
  const uint4* p = (const uint4*)As;
  int row = mrow_base + l;
  FragV f;
  f.q[0] = p[row * 5 + half];
  f.q[1] = p[row * 5 + 2 + half];
  return f.v;
}
// B fragment (32x16 bf16) from pre-transposed Wt stored [n][k], stride 40.
__device__ __forceinline__ v16bf load_frag_b40(const bf16* Bt, int ncol_base, int lane) {
  int l = lane & 15, half = lane >> 4;
  const uint4* p = (const uint4*)Bt;
  int n = ncol_base + l;
  FragV f;
  f.q[0] = p[n * 5 + half];
  f.q[1] = p[n * 5 + 2 + half];
  return f.v;
}

// ---------------------------------------------------------------------------
// Pre-pass: f32 -> bf16 bulk convert (hidden), and weight convert+transpose.
// ---------------------------------------------------------------------------
__global__ __launch_bounds__(256) void conv_bf16_kernel(
    const float* __restrict__ x, bf16* __restrict__ y) {
  size_t i = ((size_t)blockIdx.x * 256 + threadIdx.x) * 4;
  float4 v = *(const float4*)(x + i);
  v4bf o = { f2b(v.x), f2b(v.y), f2b(v.z), f2b(v.w) };
  *(v4bf*)(y + i) = o;
}

__global__ __launch_bounds__(256) void transpose_conv_kernel(
    const float* __restrict__ W, bf16* __restrict__ Wt, int K, int N) {
  __shared__ float tile[32][33];
  int tx = threadIdx.x & 31, ty = threadIdx.x >> 5;       // 32 x 8
  int n = blockIdx.x * 32 + tx;
#pragma unroll
  for (int j = 0; j < 32; j += 8)
    tile[ty + j][tx] = W[(size_t)(blockIdx.y * 32 + ty + j) * N + n];
  __syncthreads();
  int k = blockIdx.y * 32 + tx;
#pragma unroll
  for (int j = 0; j < 32; j += 8)
    Wt[(size_t)(blockIdx.x * 32 + ty + j) * K + k] = f2b(tile[tx][ty + j]);
}

// ---------------------------------------------------------------------------
// Masked mean over K tokens -> context (bf16)
// ---------------------------------------------------------------------------
__global__ __launch_bounds__(128) void context_kernel(
    const float* __restrict__ hidden, const unsigned char* __restrict__ mask,
    bf16* __restrict__ ctx) {
  int b = blockIdx.x, tid = threadIdx.x;
  __shared__ float mk[16];
  if (tid < 16) mk[tid] = mask[b * 16 + tid] ? 1.f : 0.f;
  __syncthreads();
  float n = 0.f;
#pragma unroll
  for (int k = 0; k < 16; ++k) n += mk[k];
  float inv = 1.f / fmaxf(n, 1.f);
#pragma unroll
  for (int i = 0; i < 4; ++i) {
    int d = tid + i * 128;
    float acc = 0.f;
#pragma unroll
    for (int k = 0; k < 16; ++k)
      acc += mk[k] * hidden[((size_t)b * 16 + k) * 512 + d];
    ctx[(size_t)b * 512 + d] = f2b(acc * inv);
  }
}

// ---------------------------------------------------------------------------
// Async-staged WMMA GEMM: C[M,N] = act(A[M,K] @ Wt^T + bias1 + bias2)
// A: bf16 row-major.  Wt: bf16 pre-transposed [N][K].
// Tile 128x64, BK=32, 256 thr (8 waves), double-buffered LDS slabs.
// ---------------------------------------------------------------------------
__device__ __forceinline__ void stage_slab(
    unsigned ldsA, unsigned ldsB, const bf16* A, const bf16* Wt,
    int m0, int n0, int kk, int K, int tid) {
#pragma unroll
  for (int i = 0; i < 2; ++i) {                 // A: 128 rows x 64B
    int c = i * 256 + tid;
    int row = c >> 2, cc = c & 3;
    async_b128(ldsA + row * 80 + cc * 16,
               (const char*)(A + (size_t)(m0 + row) * K + kk) + cc * 16);
  }
  {                                             // B: 64 rows x 64B
    int row = tid >> 2, cc = tid & 3;
    async_b128(ldsB + row * 80 + cc * 16,
               (const char*)(Wt + (size_t)(n0 + row) * K + kk) + cc * 16);
  }
}

template <typename OT, bool GELU>
__global__ __launch_bounds__(256) void gemm_async_kernel(
    const bf16* __restrict__ A, const bf16* __restrict__ Wt,
    const float* __restrict__ bias1, const float* __restrict__ bias2,
    OT* __restrict__ C, bf16* __restrict__ C2, int K, int N) {
  __shared__ __align__(16) bf16 As[2][128 * 40];
  __shared__ __align__(16) bf16 Bt[2][64 * 40];
  int tid = threadIdx.x;
  int wave = tid >> 5, lane = tid & 31;
  int n0 = blockIdx.x * 64;
  int m0 = blockIdx.y * 128;
  unsigned lA[2] = { lds_addr(&As[0][0]), lds_addr(&As[1][0]) };
  unsigned lB[2] = { lds_addr(&Bt[0][0]), lds_addr(&Bt[1][0]) };
  v8f c[4] = {};

  stage_slab(lA[0], lB[0], A, Wt, m0, n0, 0, K, tid);
  int nk = K >> 5;
  for (int i = 0; i < nk; ++i) {
    wait_async0();
    __syncthreads();                            // slab i ready; compute i-1 done
    if (i + 1 < nk)
      stage_slab(lA[(i + 1) & 1], lB[(i + 1) & 1], A, Wt, m0, n0, (i + 1) * 32, K, tid);
    const bf16* Ab = As[i & 1];
    const bf16* Bb = Bt[i & 1];
    v16bf a = load_frag_a40(Ab, wave * 16, lane);
#pragma unroll
    for (int nt = 0; nt < 4; ++nt) {
      v16bf b = load_frag_b40(Bb, nt * 16, lane);
      c[nt] = __builtin_amdgcn_wmma_f32_16x16x32_bf16(
          false, a, false, b, (short)0, c[nt], false, false);
    }
  }

  int l = lane & 15, half = lane >> 4;
#pragma unroll
  for (int nt = 0; nt < 4; ++nt) {
    int col = n0 + nt * 16 + l;
    float bia = (bias1 ? bias1[col] : 0.f) + (bias2 ? bias2[col] : 0.f);
#pragma unroll
    for (int r = 0; r < 8; ++r) {
      int row = m0 + wave * 16 + r + 8 * half;
      float v = c[nt][r] + bia;
      if (GELU) v = 0.5f * v * (1.f + erff(v * 0.70710678118654752f));
      C[(size_t)row * N + col] = (OT)v;
      if (C2) C2[(size_t)row * N + col] = f2b(v);
    }
  }
}

// ---------------------------------------------------------------------------
// Workspace attention (masked softmax over K tokens) + mix + LayerNorm.
// ws (f32 [B,S,D]) updated in place; also emits bf16 copy for later GEMMs.
// ---------------------------------------------------------------------------
__global__ __launch_bounds__(128) void attn1_kernel(
    const bf16* __restrict__ Qw, const bf16* __restrict__ Kw,
    const bf16* __restrict__ Vw, float* __restrict__ ws,
    bf16* __restrict__ ws16, const unsigned char* __restrict__ mask,
    const float* __restrict__ g_ws, const float* __restrict__ b_ws) {
  int b = blockIdx.x, tid = threadIdx.x;
  __shared__ bf16 q[4 * 512], kw[16 * 512], vw[16 * 512];
  __shared__ float wsr[4 * 512];
  __shared__ float probs[4 * 16];
  __shared__ float mk[16];
  const bf16* Qg = Qw + (size_t)b * 4 * 512;
  const bf16* Kg = Kw + (size_t)b * 16 * 512;
  const bf16* Vg = Vw + (size_t)b * 16 * 512;
  float* Wg = ws + (size_t)b * 4 * 512;
  bf16* Wg16 = ws16 + (size_t)b * 4 * 512;
  for (int i = tid; i < 2048; i += 128) { q[i] = Qg[i]; wsr[i] = Wg[i]; }
  for (int i = tid; i < 8192; i += 128) { kw[i] = Kg[i]; vw[i] = Vg[i]; }
  if (tid < 16) mk[tid] = mask[b * 16 + tid] ? 1.f : 0.f;
  __syncthreads();

  if (tid < 64) {                               // logits 4x16
    int s = tid >> 4, k = tid & 15;
    float acc = 0.f;
    for (int d = 0; d < 512; ++d) acc += b2f(q[s * 512 + d]) * b2f(kw[k * 512 + d]);
    acc *= 0.0441941738241592f;                 // 512^-0.5
    probs[s * 16 + k] = (mk[k] > 0.f) ? acc : -3.0e38f;
  }
  __syncthreads();
  if (tid < 4) {                                // masked softmax over K
    int s = tid;
    float m = -3.0e38f;
#pragma unroll
    for (int k = 0; k < 16; ++k) m = fmaxf(m, probs[s * 16 + k]);
    float e[16], sum = 0.f;
#pragma unroll
    for (int k = 0; k < 16; ++k) {
      e[k] = (probs[s * 16 + k] > -1.0e38f) ? __expf(probs[s * 16 + k] - m) : 0.f;
      sum += e[k];
    }
    float inv = (sum > 0.f) ? 1.f / sum : 0.f;  // any_active==false -> zeros
#pragma unroll
    for (int k = 0; k < 16; ++k) probs[s * 16 + k] = e[k] * inv;
  }
  __syncthreads();
  for (int i = tid; i < 2048; i += 128) {       // ws + attn @ Vw
    int s = i >> 9, d = i & 511;
    float acc = wsr[i];
#pragma unroll
    for (int k = 0; k < 16; ++k) acc += probs[s * 16 + k] * b2f(vw[k * 512 + d]);
    wsr[i] = acc;
  }
  __syncthreads();
  {                                             // LayerNorm, wave per row
    int wv = tid >> 5, lane = tid & 31;
    float su = 0.f, sq = 0.f;
#pragma unroll
    for (int j = 0; j < 16; ++j) {
      float x = wsr[wv * 512 + lane + 32 * j];
      su += x; sq += x * x;
    }
#pragma unroll
    for (int off = 16; off; off >>= 1) { su += __shfl_xor(su, off); sq += __shfl_xor(sq, off); }
    float mu = su * (1.f / 512.f);
    float var = sq * (1.f / 512.f) - mu * mu;
    float rs = rsqrtf(var + 1e-5f);
#pragma unroll
    for (int j = 0; j < 16; ++j) {
      int d = lane + 32 * j;
      float x = (wsr[wv * 512 + d] - mu) * rs * g_ws[d] + b_ws[d];
      Wg[wv * 512 + d] = x;
      Wg16[wv * 512 + d] = f2b(x);
    }
  }
}

// ---------------------------------------------------------------------------
// Broadcast attention (softmax over S=4) + mix -> bf16 [B,K,D]
// ---------------------------------------------------------------------------
__global__ __launch_bounds__(128) void attn2_kernel(
    const bf16* __restrict__ Qr, const bf16* __restrict__ Kr,
    const bf16* __restrict__ Vr, bf16* __restrict__ brd) {
  int b = blockIdx.x, tid = threadIdx.x;
  __shared__ bf16 q[16 * 512], kr[4 * 512], vr[4 * 512];
  __shared__ float probs[16 * 4];
  const bf16* Qg = Qr + (size_t)b * 16 * 512;
  const bf16* Kg = Kr + (size_t)b * 4 * 512;
  const bf16* Vg = Vr + (size_t)b * 4 * 512;
  bf16* Og = brd + (size_t)b * 16 * 512;
  for (int i = tid; i < 8192; i += 128) q[i] = Qg[i];
  for (int i = tid; i < 2048; i += 128) { kr[i] = Kg[i]; vr[i] = Vg[i]; }
  __syncthreads();
  if (tid < 64) {
    int k = tid >> 2, s = tid & 3;
    float acc = 0.f;
    for (int d = 0; d < 512; ++d) acc += b2f(q[k * 512 + d]) * b2f(kr[s * 512 + d]);
    probs[k * 4 + s] = acc * 0.0441941738241592f;
  }
  __syncthreads();
  if (tid < 16) {
    int k = tid;
    float p0 = probs[k * 4], p1 = probs[k * 4 + 1], p2 = probs[k * 4 + 2], p3 = probs[k * 4 + 3];
    float m = fmaxf(fmaxf(p0, p1), fmaxf(p2, p3));
    float e0 = __expf(p0 - m), e1 = __expf(p1 - m), e2 = __expf(p2 - m), e3 = __expf(p3 - m);
    float inv = 1.f / (e0 + e1 + e2 + e3);
    probs[k * 4] = e0 * inv; probs[k * 4 + 1] = e1 * inv;
    probs[k * 4 + 2] = e2 * inv; probs[k * 4 + 3] = e3 * inv;
  }
  __syncthreads();
  for (int i = tid; i < 8192; i += 128) {
    int k = i >> 9, d = i & 511;
    float acc = 0.f;
#pragma unroll
    for (int s = 0; s < 4; ++s) acc += probs[k * 4 + s] * b2f(vr[s * 512 + d]);
    Og[i] = f2b(acc);
  }
}

// ---------------------------------------------------------------------------
// Final GEMM (brd @ ro) with fused residual + LayerNorm epilogue.
// 32 rows x full N=512 per block; async-staged slabs (single-buffered).
// ---------------------------------------------------------------------------
__global__ __launch_bounds__(256) void gemm_ro_ln_kernel(
    const bf16* __restrict__ A, const bf16* __restrict__ Wt,
    const float* __restrict__ hidden, const float* __restrict__ g,
    const float* __restrict__ bta, float* __restrict__ out) {
  const int K = 512;
  __shared__ __align__(16) bf16 As[32 * 40];
  __shared__ __align__(16) bf16 Bt[512 * 40];
  __shared__ float redS[32], redQ[32];
  int tid = threadIdx.x, wave = tid >> 5, lane = tid & 31;
  int m0 = blockIdx.x * 32;
  int m_idx = wave & 1;            // 2 m-subtiles
  int ngrp = wave >> 1;            // 4 groups x 8 n-subtiles = 32
  unsigned lA = lds_addr(&As[0]);
  unsigned lB = lds_addr(&Bt[0]);
  v8f c[8] = {};

  for (int kk = 0; kk < K; kk += 32) {
    if (tid < 128) {                             // A: 32 rows x 64B
      int row = tid >> 2, cc = tid & 3;
      async_b128(lA + row * 80 + cc * 16,
                 (const char*)(A + (size_t)(m0 + row) * K + kk) + cc * 16);
    }
#pragma unroll
    for (int i = 0; i < 8; ++i) {                // B: 512 rows x 64B
      int cidx = i * 256 + tid;
      int row = cidx >> 2, cc = cidx & 3;
      async_b128(lB + row * 80 + cc * 16,
                 (const char*)(Wt + (size_t)row * K + kk) + cc * 16);
    }
    wait_async0();
    __syncthreads();
    v16bf a = load_frag_a40(As, m_idx * 16, lane);
#pragma unroll
    for (int j = 0; j < 8; ++j) {
      v16bf b = load_frag_b40(Bt, (ngrp * 8 + j) * 16, lane);
      c[j] = __builtin_amdgcn_wmma_f32_16x16x32_bf16(
          false, a, false, b, (short)0, c[j], false, false);
    }
    __syncthreads();
  }

  int l = lane & 15, half = lane >> 4;
  if (tid < 32) { redS[tid] = 0.f; redQ[tid] = 0.f; }
  __syncthreads();
#pragma unroll
  for (int j = 0; j < 8; ++j) {                  // residual add in registers
    int col = (ngrp * 8 + j) * 16 + l;
#pragma unroll
    for (int r = 0; r < 8; ++r) {
      int rl = m_idx * 16 + r + 8 * half;
      c[j][r] += hidden[(size_t)(m0 + rl) * 512 + col];
    }
  }
#pragma unroll
  for (int r = 0; r < 8; ++r) {                  // per-row stats
    float s = 0.f, sq = 0.f;
#pragma unroll
    for (int j = 0; j < 8; ++j) { float x = c[j][r]; s += x; sq += x * x; }
#pragma unroll
    for (int off = 1; off < 16; off <<= 1) { s += __shfl_xor(s, off); sq += __shfl_xor(sq, off); }
    int rl = m_idx * 16 + r + 8 * half;
    if ((lane & 15) == 0) { atomicAdd(&redS[rl], s); atomicAdd(&redQ[rl], sq); }
  }
  __syncthreads();
#pragma unroll
  for (int j = 0; j < 8; ++j) {
    int col = (ngrp * 8 + j) * 16 + l;
    float gg = g[col], bb = bta[col];
#pragma unroll
    for (int r = 0; r < 8; ++r) {
      int rl = m_idx * 16 + r + 8 * half;
      float mu = redS[rl] * (1.f / 512.f);
      float var = redQ[rl] * (1.f / 512.f) - mu * mu;
      float rs = rsqrtf(var + 1e-5f);
      out[(size_t)(m0 + rl) * 512 + col] = (c[j][r] - mu) * rs * gg + bb;
    }
  }
}

// ---------------------------------------------------------------------------
// Host-side launch pipeline
// ---------------------------------------------------------------------------
extern "C" void kernel_launch(void* const* d_in, const int* in_sizes, int n_in,
                              void* d_out, int out_size, void* d_ws, size_t ws_size,
                              hipStream_t stream) {
  const float* hidden       = (const float*)d_in[0];
  const unsigned char* mask = (const unsigned char*)d_in[1];
  const float* fallback     = (const float*)d_in[2];
  const float* gw1          = (const float*)d_in[3];
  const float* gb1          = (const float*)d_in[4];
  const float* gw2          = (const float*)d_in[5];
  const float* gb2          = (const float*)d_in[6];
  const float* wq           = (const float*)d_in[7];
  const float* wk           = (const float*)d_in[8];
  const float* wv           = (const float*)d_in[9];
  const float* rq           = (const float*)d_in[10];
  const float* rk           = (const float*)d_in[11];
  const float* rv           = (const float*)d_in[12];
  const float* ro           = (const float*)d_in[13];
  const float* g_ws         = (const float*)d_in[14];
  const float* b_ws         = (const float*)d_in[15];
  const float* g_out        = (const float*)d_in[16];
  const float* b_out        = (const float*)d_in[17];
  float* out = (float*)d_out;

  char* ws = (char*)d_ws;
  const size_t MB = (size_t)1 << 20;
  bf16*  ctx   = (bf16*)(ws + 0);          //  8 MB [B,512]
  bf16*  h1    = (bf16*)(ws + 8 * MB);     // 16 MB [B,1024]
  float* wsb   = (float*)(ws + 24 * MB);   // 64 MB [B,S,512] f32 (in-place upd)
  bf16*  ws16  = (bf16*)(ws + 88 * MB);    // 32 MB ws bf16 -> later ws_upd bf16
  bf16*  Kw    = (bf16*)(ws + 120 * MB);   // 32 MB
  bf16*  Vw    = (bf16*)(ws + 152 * MB);   // 32 MB
  bf16*  Qw    = (bf16*)(ws + 184 * MB);   // 16 MB
  bf16*  hid16 = (bf16*)(ws + 200 * MB);   // 128 MB [B*K,512]
  bf16*  Qr    = (bf16*)(ws + 120 * MB);   // 64 MB (reuses Kw+Vw)
  bf16*  Kr    = (bf16*)(ws + 184 * MB);   // 16 MB (reuses Qw)
  bf16*  Vr    = (bf16*)(ws + 0);          // 16 MB (reuses ctx+h1)
  bf16*  brd   = (bf16*)(ws + 24 * MB);    // 64 MB (reuses wsb)
  char*  wbase = ws + 328 * MB;            // transposed bf16 weights (~8.5 MB)
  bf16* gw1t = (bf16*)(wbase);                       // 1 MB   [1024][512]
  bf16* gw2t = (bf16*)(wbase + 1 * MB);              // 4 MB   [2048][1024]
  bf16* wqt  = (bf16*)(wbase + 5 * MB);              // 512 KB [512][512]
  bf16* wkt  = (bf16*)(wbase + 5 * MB + 512 * 1024);
  bf16* wvt  = (bf16*)(wbase + 6 * MB);
  bf16* rqt  = (bf16*)(wbase + 6 * MB + 512 * 1024);
  bf16* rkt  = (bf16*)(wbase + 7 * MB);
  bf16* rvt  = (bf16*)(wbase + 7 * MB + 512 * 1024);
  bf16* rot  = (bf16*)(wbase + 8 * MB);

  // 0) one-time converts: hidden -> bf16; weights -> bf16 transposed [N][K]
  conv_bf16_kernel<<<65536, 256, 0, stream>>>(hidden, hid16);
  transpose_conv_kernel<<<dim3(32, 16), 256, 0, stream>>>(gw1, gw1t, 512, 1024);
  transpose_conv_kernel<<<dim3(64, 32), 256, 0, stream>>>(gw2, gw2t, 1024, 2048);
  transpose_conv_kernel<<<dim3(16, 16), 256, 0, stream>>>(wq, wqt, 512, 512);
  transpose_conv_kernel<<<dim3(16, 16), 256, 0, stream>>>(wk, wkt, 512, 512);
  transpose_conv_kernel<<<dim3(16, 16), 256, 0, stream>>>(wv, wvt, 512, 512);
  transpose_conv_kernel<<<dim3(16, 16), 256, 0, stream>>>(rq, rqt, 512, 512);
  transpose_conv_kernel<<<dim3(16, 16), 256, 0, stream>>>(rk, rkt, 512, 512);
  transpose_conv_kernel<<<dim3(16, 16), 256, 0, stream>>>(rv, rvt, 512, 512);
  transpose_conv_kernel<<<dim3(16, 16), 256, 0, stream>>>(ro, rot, 512, 512);
  // 1) masked mean -> context
  context_kernel<<<8192, 128, 0, stream>>>(hidden, mask, ctx);
  // 2) h1 = gelu(ctx @ gw1 + gb1)           M=8192  K=512  N=1024
  gemm_async_kernel<bf16, true><<<dim3(16, 64), 256, 0, stream>>>(
      ctx, gw1t, gb1, nullptr, h1, (bf16*)nullptr, 512, 1024);
  // 3) ws = h1 @ gw2 + gb2 + fallback       M=8192  K=1024 N=2048 (f32 + bf16)
  gemm_async_kernel<float, false><<<dim3(32, 64), 256, 0, stream>>>(
      h1, gw2t, gb2, fallback, wsb, ws16, 1024, 2048);
  // 4) Kw / Vw = hidden @ wk/wv             M=131072 K=512 N=512
  gemm_async_kernel<bf16, false><<<dim3(8, 1024), 256, 0, stream>>>(
      hid16, wkt, nullptr, nullptr, Kw, (bf16*)nullptr, 512, 512);
  gemm_async_kernel<bf16, false><<<dim3(8, 1024), 256, 0, stream>>>(
      hid16, wvt, nullptr, nullptr, Vw, (bf16*)nullptr, 512, 512);
  // 5) Qw = ws @ wq                         M=32768 K=512 N=512
  gemm_async_kernel<bf16, false><<<dim3(8, 256), 256, 0, stream>>>(
      ws16, wqt, nullptr, nullptr, Qw, (bf16*)nullptr, 512, 512);
  // 6) workspace attention + LN (wsb -> ws_upd in place, + bf16 copy)
  attn1_kernel<<<8192, 128, 0, stream>>>(Qw, Kw, Vw, wsb, ws16, mask, g_ws, b_ws);
  // 7) Qr = hidden @ rq ; Kr/Vr = ws_upd @ rk/rv
  gemm_async_kernel<bf16, false><<<dim3(8, 1024), 256, 0, stream>>>(
      hid16, rqt, nullptr, nullptr, Qr, (bf16*)nullptr, 512, 512);
  gemm_async_kernel<bf16, false><<<dim3(8, 256), 256, 0, stream>>>(
      ws16, rkt, nullptr, nullptr, Kr, (bf16*)nullptr, 512, 512);
  gemm_async_kernel<bf16, false><<<dim3(8, 256), 256, 0, stream>>>(
      ws16, rvt, nullptr, nullptr, Vr, (bf16*)nullptr, 512, 512);
  // 8) broadcast attention
  attn2_kernel<<<8192, 128, 0, stream>>>(Qr, Kr, Vr, brd);
  // 9) out = LN(hidden + brd @ ro)
  gemm_ro_ln_kernel<<<4096, 256, 0, stream>>>(brd, rot, hidden, g_out, b_out, out);
}
